// QwenFeedForward_adaptive_56693568307570
// MI455X (gfx1250) — matrix operand order repaired and checked by reading, not compiled, not run on hardware
//
#include <hip/hip_runtime.h>
#include <cstdint>
#include <cstddef>

// ---------------------------------------------------------------------------
// Qwen adaptive FFN for MI455X (gfx1250, wave32, WMMA + TDM).
//   out = (gelu(x@W1^T + b1) * prefix_mask) @ W2^T + b2 ;  p_soft = softmax(lw)
// Memory-bound problem =>
//   1) one elementwise pass converts x/W1/W2 to bf16 in workspace (done once,
//      not once-per-tile; W1b+W2b = 67MB stay resident in the 192MB L2),
//   2) both GEMMs run v_wmma_f32_16x16x32_bf16 with ALL tile staging done by
//      the Tensor Data Mover (tensor_load_to_lds, LDS row padding), double
//      buffered against the WMMA stream, so the main loops have near-zero
//      per-thread staging VALU,
//   3) width routing = whole-tile skips in GEMM1 + truncated K loop in GEMM2.
// ---------------------------------------------------------------------------

#define DIMC  2048
#define INNER 8192
#define MTOT  8192   // B*L = 4*2048
#define BM    128
#define BN    128
#define BK    64
#define LDSD  72     // padded LDS row stride in bf16 elems (144B, 16B-aligned)

typedef __attribute__((ext_vector_type(16))) __bf16        v16bf;
typedef __attribute__((ext_vector_type(8)))  float         v8f;
typedef __attribute__((ext_vector_type(4)))  float         v4f;
typedef __attribute__((ext_vector_type(4)))  unsigned int  v4u;
typedef __attribute__((ext_vector_type(4)))  int           v4i;
typedef __attribute__((ext_vector_type(8)))  int           v8i;

union FragU { v4u q[2]; v16bf v; };

#if __has_builtin(__builtin_amdgcn_tensor_load_to_lds) && \
    __has_builtin(__builtin_amdgcn_s_wait_tensorcnt)
#define USE_TDM 1
#else
#define USE_TDM 0
#endif

__device__ __forceinline__ unsigned int pk2(float a, float b) {
  unsigned int ua = __float_as_uint(a);
  ua += 0x7FFFu + ((ua >> 16) & 1u);          // RNE
  unsigned int ub = __float_as_uint(b);
  ub += 0x7FFFu + ((ub >> 16) & 1u);
  return (ua >> 16) | (ub & 0xFFFF0000u);
}
__device__ __forceinline__ unsigned short f2bf(float f) {
  unsigned int u = __float_as_uint(f);
  u += 0x7FFFu + ((u >> 16) & 1u);
  return (unsigned short)(u >> 16);
}

// ---------------------------------------------------------------------------
// TDM: 2D bf16 tile (tile_w halfs x tile_h rows) global -> LDS, row-padded
// (pad 4 DWORDs after every 32 DWORDs: 64-half rows -> 72-half LDS stride).
// Issued by one wave (TDM ignores EXEC); completion via s_wait_tensorcnt.
// ---------------------------------------------------------------------------
#if USE_TDM
__device__ __forceinline__ void tdm_load_tile_bf16(
    unsigned lds_off, const void* gptr,
    unsigned tile_w, unsigned tile_h,
    unsigned tensor_w, unsigned tensor_h, unsigned row_stride_elems) {
  unsigned long long ga = (unsigned long long)(size_t)gptr;

  v4u g0;
  g0.x = 1u;                                        // count=1 valid user D#
  g0.y = lds_off;                                   // lds_addr  bits[63:32]
  g0.z = (unsigned)(ga & 0xFFFFFFFFull);            // global_addr[31:0]
  g0.w = (unsigned)((ga >> 32) & 0x01FFFFFFull)     // global_addr[56:32]
       | (2u << 30);                                // type=2 ("image")

  v8i g1;
  g1[0] = (int)((1u << 16)        // data_size = 1 -> 2 bytes
              | (1u << 20)        // pad_enable
              | (4u << 22)        // pad_interval: code 4 -> every 32 DWORDs
              | (3u << 25));      // pad_amount:   code 3 -> 4 DWORDs (8 halfs)
  g1[1] = (int)((tensor_w & 0xFFFFu) << 16);          // tensor_dim0[15:0]
  g1[2] = (int)((tensor_w >> 16) | ((tensor_h & 0xFFFFu) << 16));
  g1[3] = (int)((tensor_h >> 16) | (tile_w << 16));   // tile_dim0
  g1[4] = (int)(tile_h & 0xFFFFu);                    // tile_dim1 (tile_dim2=0)
  g1[5] = (int)row_stride_elems;                      // tensor_dim0_stride[31:0]
  g1[6] = 0;
  g1[7] = 0;

  const v4i z4 = {0, 0, 0, 0};
#if defined(__clang_major__) && (__clang_major__ >= 23)
  const v8i z8 = {0, 0, 0, 0, 0, 0, 0, 0};
  __builtin_amdgcn_tensor_load_to_lds(g0, g1, z4, z4, z8, 0);
#else
  __builtin_amdgcn_tensor_load_to_lds(g0, g1, z4, z4, 0);
#endif
}
#endif

// ---------------------------------------------------------------------------
// Elementwise f32 -> bf16 (one pass; 8 elems/thread).
// ---------------------------------------------------------------------------
__global__ __launch_bounds__(256) void cvt_bf16(const float* __restrict__ src,
                                                unsigned short* __restrict__ dst,
                                                int n8 /* n/8 */) {
  int i = blockIdx.x * 256 + threadIdx.x;
  if (i < n8) {
    const v4f* s = (const v4f*)(src + (size_t)i * 8);
    __builtin_prefetch(src + (size_t)i * 8 + 16384, 0, 1);
    v4f f0 = s[0], f1 = s[1];
    v4u r = {pk2(f0.x, f0.y), pk2(f0.z, f0.w), pk2(f1.x, f1.y), pk2(f1.z, f1.w)};
    *(v4u*)(dst + (size_t)i * 8) = r;
  }
}

// ---------------------------------------------------------------------------
// Router: softmax over [4,4] logits + hard width per batch row.
// ---------------------------------------------------------------------------
__global__ void ffn_router(const float* __restrict__ logits,
                           float* __restrict__ p_out,
                           int*   __restrict__ widths) {
  int r = threadIdx.x;
  if (r < 4) {
    float l0 = logits[r * 4 + 0], l1 = logits[r * 4 + 1];
    float l2 = logits[r * 4 + 2], l3 = logits[r * 4 + 3];
    float m  = fmaxf(fmaxf(l0, l1), fmaxf(l2, l3));
    float e0 = __expf(l0 - m), e1 = __expf(l1 - m);
    float e2 = __expf(l2 - m), e3 = __expf(l3 - m);
    float inv = 1.0f / (e0 + e1 + e2 + e3);
    p_out[r * 4 + 0] = e0 * inv; p_out[r * 4 + 1] = e1 * inv;
    p_out[r * 4 + 2] = e2 * inv; p_out[r * 4 + 3] = e3 * inv;
    int idx = 0; float best = l0;                 // first-max == jnp.argmax
    if (l1 > best) { best = l1; idx = 1; }
    if (l2 > best) { best = l2; idx = 2; }
    if (l3 > best) { best = l3; idx = 3; }
    widths[r] = (idx + 1) * (INNER / 4);
  }
}

// ---------------------------------------------------------------------------
// Shared GEMM machinery (bf16 A/B tiles, 128x128 block, BK=64).
// ---------------------------------------------------------------------------
#define STAGE_TILE(ldsArr, gsrc, rowStride, k0)                                  \
  do {                                                                           \
    const int _row = tid >> 1, _half = tid & 1;                                  \
    const v4u* _s = (const v4u*)((gsrc) + (size_t)(_row) * (rowStride) + (k0) +  \
                                 _half * 32);                                    \
    v4u* _d = (v4u*)&(ldsArr)[_row * LDSD + _half * 32];                         \
    _d[0] = _s[0]; _d[1] = _s[1]; _d[2] = _s[2]; _d[3] = _s[3];                  \
  } while (0)

#define COMPUTE_STEP(pa, pb)                                                     \
  do {                                                                           \
    _Pragma("unroll")                                                            \
    for (int kk = 0; kk < 2; ++kk) {                                             \
      v16bf a[4], b[2];                                                          \
      _Pragma("unroll")                                                          \
      for (int ms = 0; ms < 4; ++ms) {        /* A: lane = M row, regs = K */    \
        int row = wm * 64 + ms * 16 + lm;                                        \
        FragU u;                                                                 \
        u.q[0] = *(const v4u*)&(pa)[row * LDSD + kk * 32 + kg * 8];              \
        u.q[1] = *(const v4u*)&(pa)[row * LDSD + kk * 32 + 16 + kg * 8];         \
        a[ms] = u.v;                                                             \
      }                                                                          \
      _Pragma("unroll")                                                          \
      for (int ns = 0; ns < 2; ++ns) {        /* B: lane = N col */              \
        int col = wn * 32 + ns * 16 + lm;                                        \
        FragU u;                                                                 \
        u.q[0] = *(const v4u*)&(pb)[col * LDSD + kk * 32 + kg * 8];              \
        u.q[1] = *(const v4u*)&(pb)[col * LDSD + kk * 32 + 16 + kg * 8];         \
        b[ns] = u.v;                                                             \
      }                                                                          \
      _Pragma("unroll")                                                          \
      for (int ms = 0; ms < 4; ++ms)                                             \
        _Pragma("unroll")                                                        \
        for (int ns = 0; ns < 2; ++ns)                                           \
          acc[ms][ns] = __builtin_amdgcn_wmma_f32_16x16x32_bf16(                 \
              false, a[ms], false, b[ns], (short)0, acc[ms][ns], false, false);  \
    }                                                                            \
  } while (0)

// ---------------------------------------------------------------------------
// GEMM1: h = gelu(xb @ W1b^T + b1), tile-skipped beyond width[batch].
//   xb [MTOT, DIMC] bf16   W1b [INNER, DIMC] bf16   h [MTOT, INNER] bf16
// ---------------------------------------------------------------------------
__global__ __launch_bounds__(256) void ffn_gemm1(
    const unsigned short* __restrict__ xb,
    const unsigned short* __restrict__ W1b,
    const float* __restrict__ b1,
    const int*   __restrict__ widths,
    unsigned short* __restrict__ h) {
  __shared__ unsigned short sA[2][BM * LDSD];
  __shared__ unsigned short sB[2][BN * LDSD];

  const int m0 = blockIdx.y * BM;
  const int n0 = blockIdx.x * BN;
  const int width = widths[m0 >> 11];        // batch = m0 / 2048
  if (n0 >= width) return;                   // fully-masked tile: never needed

  const int tid  = threadIdx.x;
  const int lane = tid & 31;
  const int wave = tid >> 5;
  const int wm = wave >> 2, wn = wave & 3;   // 2x4 wave grid over 128x128
  const int lm = lane & 15, kg = lane >> 4;

  v8f acc[4][2];
  const v8f vzero = {0.f, 0.f, 0.f, 0.f, 0.f, 0.f, 0.f, 0.f};
  #pragma unroll
  for (int i = 0; i < 4; ++i)
    #pragma unroll
    for (int j = 0; j < 2; ++j) acc[i][j] = vzero;

  auto stage = [&](int buf, int k0) {
#if USE_TDM
    if (wave == 0) {
      tdm_load_tile_bf16((unsigned)(size_t)&sA[buf][0],
                         xb + (size_t)m0 * DIMC + k0, BK, BM, DIMC, MTOT, DIMC);
      tdm_load_tile_bf16((unsigned)(size_t)&sB[buf][0],
                         W1b + (size_t)n0 * DIMC + k0, BK, BN, DIMC, INNER, DIMC);
    }
#else
    STAGE_TILE(sA[buf], xb + (size_t)m0 * DIMC, DIMC, k0);
    STAGE_TILE(sB[buf], W1b + (size_t)n0 * DIMC, DIMC, k0);
#endif
  };

  stage(0, 0);
#if USE_TDM
  if (wave == 0) __builtin_amdgcn_s_wait_tensorcnt(0);
#endif
  __syncthreads();

  const int KT = DIMC / BK;                  // 32
  int buf = 0;
  for (int kt = 0; kt < KT; ++kt) {
    if (kt + 1 < KT) stage(buf ^ 1, (kt + 1) * BK);
    COMPUTE_STEP(sA[buf], sB[buf]);          // overlaps the in-flight TDM
#if USE_TDM
    if (wave == 0) __builtin_amdgcn_s_wait_tensorcnt(0);
#endif
    __syncthreads();
    buf ^= 1;
  }

  // epilogue: bias + x*sigmoid(1.702x), store bf16
  #pragma unroll
  for (int ns = 0; ns < 2; ++ns) {
    const int col  = n0 + wn * 32 + ns * 16 + lm;
    const float bias = b1[col];
    #pragma unroll
    for (int ms = 0; ms < 4; ++ms) {
      const int rbase = m0 + wm * 64 + ms * 16 + kg * 8;
      #pragma unroll
      for (int i = 0; i < 8; ++i) {
        float v = acc[ms][ns][i] + bias;
        float g = v * (1.0f / (1.0f + __expf(-1.702f * v)));
        h[(size_t)(rbase + i) * INNER + col] = f2bf(g);
      }
    }
  }
}

// ---------------------------------------------------------------------------
// GEMM2: out = h @ W2b^T + b2 ; K loop truncated at width[batch].
//   h [MTOT, INNER] bf16   W2b [DIMC, INNER] bf16   out [MTOT, DIMC] f32
// ---------------------------------------------------------------------------
__global__ __launch_bounds__(256) void ffn_gemm2(
    const unsigned short* __restrict__ h,
    const unsigned short* __restrict__ W2b,
    const float* __restrict__ b2,
    const int*   __restrict__ widths,
    float* __restrict__ out) {
  __shared__ unsigned short sA[2][BM * LDSD];
  __shared__ unsigned short sB[2][BN * LDSD];

  const int m0 = blockIdx.y * BM;
  const int n0 = blockIdx.x * BN;
  const int kmax = widths[m0 >> 11];         // truncated reduction

  const int tid  = threadIdx.x;
  const int lane = tid & 31;
  const int wave = tid >> 5;
  const int wm = wave >> 2, wn = wave & 3;
  const int lm = lane & 15, kg = lane >> 4;

  v8f acc[4][2];
  const v8f vzero = {0.f, 0.f, 0.f, 0.f, 0.f, 0.f, 0.f, 0.f};
  #pragma unroll
  for (int i = 0; i < 4; ++i)
    #pragma unroll
    for (int j = 0; j < 2; ++j) acc[i][j] = vzero;

  auto stage = [&](int buf, int k0) {
#if USE_TDM
    if (wave == 0) {
      tdm_load_tile_bf16((unsigned)(size_t)&sA[buf][0],
                         h + (size_t)m0 * INNER + k0, BK, BM, INNER, MTOT, INNER);
      tdm_load_tile_bf16((unsigned)(size_t)&sB[buf][0],
                         W2b + (size_t)n0 * INNER + k0, BK, BN, INNER, DIMC, INNER);
    }
#else
    STAGE_TILE(sA[buf], h + (size_t)m0 * INNER, INNER, k0);
    STAGE_TILE(sB[buf], W2b + (size_t)n0 * INNER, INNER, k0);
#endif
  };

  stage(0, 0);
#if USE_TDM
  if (wave == 0) __builtin_amdgcn_s_wait_tensorcnt(0);
#endif
  __syncthreads();

  const int KT = kmax / BK;                  // 32..128 by routed width
  int buf = 0;
  for (int kt = 0; kt < KT; ++kt) {
    if (kt + 1 < KT) stage(buf ^ 1, (kt + 1) * BK);
    COMPUTE_STEP(sA[buf], sB[buf]);
#if USE_TDM
    if (wave == 0) __builtin_amdgcn_s_wait_tensorcnt(0);
#endif
    __syncthreads();
    buf ^= 1;
  }

  #pragma unroll
  for (int ns = 0; ns < 2; ++ns) {
    const int col  = n0 + wn * 32 + ns * 16 + lm;
    const float bias = b2[col];
    #pragma unroll
    for (int ms = 0; ms < 4; ++ms) {
      const int rbase = m0 + wm * 64 + ms * 16 + kg * 8;
      #pragma unroll
      for (int i = 0; i < 8; ++i)
        out[(size_t)(rbase + i) * DIMC + col] = acc[ms][ns][i] + bias;
    }
  }
}

// ---------------------------------------------------------------------------
extern "C" void kernel_launch(void* const* d_in, const int* in_sizes, int n_in,
                              void* d_out, int out_size, void* d_ws, size_t ws_size,
                              hipStream_t stream) {
  const float* x  = (const float*)d_in[0];   // hidden_states [4,2048,2048]
  const float* lw = (const float*)d_in[1];   // logit_width   [4,4]
  const float* W1 = (const float*)d_in[2];   // [8192,2048]
  const float* b1 = (const float*)d_in[3];   // [8192]
  const float* W2 = (const float*)d_in[4];   // [2048,8192]
  const float* b2 = (const float*)d_in[5];   // [2048]

  float* out    = (float*)d_out;                       // [8192,2048]
  float* p_soft = out + (size_t)MTOT * DIMC;           // 16 floats (tuple tail)

  // workspace layout (256B-aligned sections)
  char* ws = (char*)d_ws;
  int* widths = (int*)ws;                                       // 16 B
  unsigned short* xb  = (unsigned short*)(ws + 256);            // 33.5 MB
  unsigned short* W1b = xb  + (size_t)MTOT  * DIMC;             // 33.5 MB
  unsigned short* W2b = W1b + (size_t)INNER * DIMC;             // 33.5 MB
  unsigned short* hb  = W2b + (size_t)DIMC  * INNER;            // 128 MB

  ffn_router<<<1, 32, 0, stream>>>(lw, p_soft, widths);

  const int n8 = (MTOT * DIMC) / 8;          // same element count for all three
  cvt_bf16<<<n8 / 256, 256, 0, stream>>>(x,  xb,  n8);
  cvt_bf16<<<n8 / 256, 256, 0, stream>>>(W1, W1b, n8);
  cvt_bf16<<<n8 / 256, 256, 0, stream>>>(W2, W2b, n8);

  dim3 g1(INNER / BN, MTOT / BM);   // (64, 64)
  ffn_gemm1<<<g1, 256, 0, stream>>>(xb, W1b, b1, widths, hb);

  dim3 g2(DIMC / BN, MTOT / BM);    // (16, 64)
  ffn_gemm2<<<g2, 256, 0, stream>>>(hb, W2b, b2, widths, out);
}
